// DCTLayer_45689862095529
// MI455X (gfx1250) — matrix operand order repaired and checked by reading, not compiled, not run on hardware
//
#include <hip/hip_runtime.h>
#include <hip/hip_bf16.h>

typedef __attribute__((ext_vector_type(16))) _Float16 v16h;
typedef __attribute__((ext_vector_type(8)))  _Float16 v8h;
typedef __attribute__((ext_vector_type(8)))  float    v8f;

#define IMG_H   224
#define IMG_W   224
#define PIX     (IMG_H * IMG_W)       // 50176
#define NFREQ   32                    // 28 output freqs padded to 32 (2 WMMA tiles)
#define OUTN    28
#define TWO_PI_OVER_N 0.02804930024f  // 2*pi/224

// ---------------- LDS layout (in _Float16 units, all 32B aligned) ----------
// gray_t : [224][224]  transposed grayscale (w-major)          50176
// costab : [32][224]   cos(2*pi*u*h/224), u-major               7168
// sintab : [32][224]                                            7168
// a1tab  : [32][224]   C*G   (f16, u-major)                     7168
// a2tab  : [32][224]   S*G                                      7168
// total = 78848 halves = 157696 bytes
#define LDS_HALVES 78848

// ---- fragment loaders (per-lane LDS gathers matching CDNA5 WMMA layouts) --

// A-matrix 16x32 f16: lane l holds row m = base+(l&15);
// halves j=0..7  -> K = kb + (l>>4)*8 + j        (contiguous 16B chunk)
// halves j=8..15 -> K = kb + 16 + (l>>4)*8 + j-8 (contiguous 16B chunk)
__device__ __forceinline__ v16h load_fragA(const _Float16* tab, int mbase,
                                           int kb, int lane) {
    const int row = mbase + (lane & 15);
    const int off = row * IMG_H + kb + ((lane >> 4) << 3);
    v8h lo = *(const v8h*)(tab + off);
    v8h hi = *(const v8h*)(tab + off + 16);
    return __builtin_shufflevector(lo, hi, 0, 1, 2, 3, 4, 5, 6, 7,
                                            8, 9, 10, 11, 12, 13, 14, 15);
}

// B-matrix 32x16 f16 from a column-major table colmaj[n*224 + k]:
// lane l holds column n = nbase+(l&15); halves s=0..7 -> K = kb+(l>>4)*8+s,
// halves s=8..15 -> K = kb+16+(l>>4)*8+s-8 (two 16B loads)
__device__ __forceinline__ v16h load_fragB(const _Float16* colmaj, int nbase,
                                           int kb, int lane) {
    const int off = (nbase + (lane & 15)) * IMG_H + kb + ((lane >> 4) << 3);
    v8h lo = *(const v8h*)(colmaj + off);
    v8h hi = *(const v8h*)(colmaj + off + 16);
    return __builtin_shufflevector(lo, hi, 0, 1, 2, 3, 4, 5, 6, 7,
                                            8, 9, 10, 11, 12, 13, 14, 15);
}

__global__ __launch_bounds__(256)
void dct28_wmma_kernel(const float* __restrict__ x, float* __restrict__ out) {
    extern __shared__ _Float16 smem[];
    _Float16* gray_t = smem;                     // [w][h]
    _Float16* costab = smem + PIX;               // [u][h]
    _Float16* sintab = costab + NFREQ * IMG_H;
    _Float16* a1tab  = sintab + NFREQ * IMG_H;   // [u][w]
    _Float16* a2tab  = a1tab  + NFREQ * IMG_H;

    const int b    = blockIdx.x;
    const int tid  = threadIdx.x;
    const int lane = tid & 31;
    const int wave = tid >> 5;

    // ---- Stage 0: fused grayscale + f16 conversion + transposed LDS store
    const float* img = x + (size_t)b * 3 * PIX;
    const float4* r4 = (const float4*)(img);
    const float4* g4 = (const float4*)(img + PIX);
    const float4* b4 = (const float4*)(img + 2 * PIX);
    for (int i = tid; i < PIX / 4; i += 256) {     // 49 coalesced float4 iters
        const float4 r = r4[i], g = g4[i], bl = b4[i];
        float gr[4];
        gr[0] = 0.2989f * r.x + 0.587f * g.x + 0.114f * bl.x;
        gr[1] = 0.2989f * r.y + 0.587f * g.y + 0.114f * bl.y;
        gr[2] = 0.2989f * r.z + 0.587f * g.z + 0.114f * bl.z;
        gr[3] = 0.2989f * r.w + 0.587f * g.w + 0.114f * bl.w;
        const int p = i * 4;
#pragma unroll
        for (int e = 0; e < 4; ++e) {
            const int pp = p + e;
            const int h = pp / IMG_W;
            const int w = pp - h * IMG_W;
            gray_t[w * IMG_H + h] = (_Float16)gr[e];   // transposed
        }
    }

    // ---- Stage 0b: DFT basis tables  cos/sin(2*pi*u*h/224), u = 0..31
    for (int i = tid; i < NFREQ * IMG_H; i += 256) {
        const int u  = i / IMG_H;
        const int h  = i - u * IMG_H;
        const float t = (float)((u * h) % IMG_H) * TWO_PI_OVER_N;
        costab[i] = (_Float16)__cosf(t);
        sintab[i] = (_Float16)__sinf(t);
    }
    __syncthreads();

    // ---- Stage 1: A1 = C*G, A2 = S*G   (32x224, K = 224 rows of gray)
    // 28 output tiles: 2 M-tiles (u) x 14 N-tiles (w); round-robin on 8 waves
    for (int t = wave; t < 28; t += 8) {
        const int ubase = (t & 1) * 16;
        const int wbase = (t >> 1) * 16;
        v8f acc1 = {};
        v8f acc2 = {};
#pragma unroll
        for (int kb = 0; kb < 7; ++kb) {
            const int hb = kb * 32;
            const v16h bg = load_fragB(gray_t, wbase, hb, lane); // G[h,w] tile
            const v16h ac = load_fragA(costab, ubase, hb, lane);
            const v16h as = load_fragA(sintab, ubase, hb, lane);
            acc1 = __builtin_amdgcn_wmma_f32_16x16x32_f16(
                false, ac, false, bg, (short)0, acc1, false, false);
            acc2 = __builtin_amdgcn_wmma_f32_16x16x32_f16(
                false, as, false, bg, (short)0, acc2, false, false);
        }
        // D layout: lane holds N-col = lane&15; VGPR r -> M row = (lane>=16?8:0)+r
        const int ur = ubase + ((lane >> 4) << 3);
        const int wc = wbase + (lane & 15);
#pragma unroll
        for (int r = 0; r < 8; ++r) {
            a1tab[(ur + r) * IMG_H + wc] = (_Float16)acc1[r];
            a2tab[(ur + r) * IMG_H + wc] = (_Float16)acc2[r];
        }
    }
    __syncthreads();

    // ---- Stage 2: out = A1*C^T - A2*S^T   (32x32, keep 28x28 corner)
    if (wave < 4) {   // wave-uniform: EXEC all ones inside WMMA region
        const int ubase = (wave & 1) * 16;
        const int vbase = (wave >> 1) * 16;
        v8f acc1 = {};
        v8f acc2 = {};
#pragma unroll
        for (int kb = 0; kb < 7; ++kb) {
            const int wb = kb * 32;
            const v16h a1 = load_fragA(a1tab, ubase, wb, lane);
            const v16h a2 = load_fragA(a2tab, ubase, wb, lane);
            // B[k=w, n=v] = costab[v*224 + w]  (table is already "column major")
            const v16h bc = load_fragB(costab, vbase, wb, lane);
            const v16h bs = load_fragB(sintab, vbase, wb, lane);
            acc1 = __builtin_amdgcn_wmma_f32_16x16x32_f16(
                false, a1, false, bc, (short)0, acc1, false, false);
            acc2 = __builtin_amdgcn_wmma_f32_16x16x32_f16(
                false, a2, false, bs, (short)0, acc2, false, false);
        }
        const int ur = ubase + ((lane >> 4) << 3);
        const int v  = vbase + (lane & 15);
        if (v < OUTN) {
            float* ob = out + (size_t)b * OUTN * OUTN;
#pragma unroll
            for (int r = 0; r < 8; ++r) {
                const int u = ur + r;
                if (u < OUTN) ob[u * OUTN + v] = acc1[r] - acc2[r];
            }
        }
    }
}

extern "C" void kernel_launch(void* const* d_in, const int* in_sizes, int n_in,
                              void* d_out, int out_size, void* d_ws, size_t ws_size,
                              hipStream_t stream) {
    (void)in_sizes; (void)n_in; (void)d_ws; (void)ws_size; (void)out_size;
    const float* x = (const float*)d_in[0];
    float* out = (float*)d_out;

    const size_t lds_bytes = (size_t)LDS_HALVES * sizeof(_Float16); // 157696 B
    // opt in to >64KB dynamic LDS where the runtime enforces a default cap
    hipFuncSetAttribute((const void*)dct28_wmma_kernel,
                        hipFuncAttributeMaxDynamicSharedMemorySize,
                        (int)lds_bytes);

    dct28_wmma_kernel<<<512, 256, lds_bytes, stream>>>(x, out);
}
